// MultiheadAttentionalAggregation_56014963474967
// MI455X (gfx1250) — matrix operand order repaired and checked by reading, not compiled, not run on hardware
//
#include <hip/hip_runtime.h>
#include <hip/hip_bf16.h>

typedef __attribute__((ext_vector_type(16))) __bf16 v16bf;
typedef __attribute__((ext_vector_type(8)))  float  v8f;

union Frag {
    v16bf v;
    uint4 q[2];
};

__device__ __forceinline__ void atomicMaxF(float* addr, float v) {
    // Monotone-toward-max under IEEE ordering for both sign paths.
    if (v >= 0.0f) atomicMax((int*)addr, __float_as_int(v));
    else           atomicMin((unsigned int*)addr, __float_as_uint(v));
}

// ---------------- init: zero out + s, m = -inf --------------------------
__global__ void init_k(float* __restrict__ out, float* __restrict__ m,
                       float* __restrict__ s, int G) {
    int idx = blockIdx.x * 256 + threadIdx.x;
    if (idx < G * 256) out[idx] = 0.0f;
    if (idx < G * 8) {
        m[idx] = __uint_as_float(0xff800000u);   // -inf
        s[idx] = 0.0f;
    }
}

// ---------------- repack Wn[H,C,D] -> WbT[col=h*32+d][k=c] bf16 ---------
__global__ void wcvt_k(const float* __restrict__ Wn, __bf16* __restrict__ WbT) {
    int idx = blockIdx.x * 256 + threadIdx.x;     // 65536 total
    int col = idx >> 8;                            // 0..255  (h*32+d)
    int c   = idx & 255;                           // 0..255  (K)
    int h = col >> 5, d = col & 31;
    WbT[col * 256 + c] = (__bf16)Wn[h * 8192 + c * 32 + d];
}

// ---------------- gate = x @ Wg^T, segment max --------------------------
__global__ void gate_k(const float* __restrict__ x,
                       const long long* __restrict__ batch,
                       const float* __restrict__ Wg,
                       float* __restrict__ gate, float* __restrict__ m, int N) {
    __shared__ float wg_lds[8 * 256];
    int tid = threadIdx.x;
    #pragma unroll
    for (int i = 0; i < 8; ++i) wg_lds[tid + i * 256] = Wg[tid + i * 256];
    __syncthreads();

    int lane = tid & 31, w = tid >> 5;
    #pragma unroll
    for (int it = 0; it < 4; ++it) {
        int n = blockIdx.x * 32 + it * 8 + w;
        bool valid = n < N;
        int nn = valid ? n : N - 1;
        const float4* xp = (const float4*)x + (size_t)nn * 64 + (lane << 1);
        float4 xa = xp[0], xb = xp[1];
        float xr[8] = {xa.x, xa.y, xa.z, xa.w, xb.x, xb.y, xb.z, xb.w};
        float sums[8];
        #pragma unroll
        for (int h = 0; h < 8; ++h) {
            const float* wrow = &wg_lds[h * 256 + (lane << 3)];
            float a = 0.0f;
            #pragma unroll
            for (int i = 0; i < 8; ++i) a += xr[i] * wrow[i];
            sums[h] = a;
        }
        #pragma unroll
        for (int h = 0; h < 8; ++h) {
            #pragma unroll
            for (int off = 16; off > 0; off >>= 1)
                sums[h] += __shfl_xor(sums[h], off, 32);
        }
        if (valid && lane < 8) {
            float gt = sums[0];
            #pragma unroll
            for (int h = 1; h < 8; ++h) if (lane == h) gt = sums[h];
            gate[(size_t)n * 8 + lane] = gt;
            int g = (int)batch[n];
            atomicMaxF(&m[g * 8 + lane], gt);
        }
    }
}

// ---------------- e = exp(gate - m), segment sum (in-place) -------------
__global__ void exp_k(const long long* __restrict__ batch,
                      float* __restrict__ gate, const float* __restrict__ m,
                      float* __restrict__ s, int N) {
    int idx = blockIdx.x * 256 + threadIdx.x;
    if (idx >= N * 8) return;
    int n = idx >> 3, h = idx & 7;
    int g = (int)batch[n];
    float e = __expf(gate[idx] - m[g * 8 + h]);
    gate[idx] = e;
    atomicAdd(&s[g * 8 + h], e);
}

// ---------------- main: bf16 WMMA projection + weighted segment sum -----
#define ROWS 64
#define XSTRIDE 264   // halves, padded: conflict-free ds_load_b128 A-fragments

__global__ void main_k(const float* __restrict__ x,
                       const long long* __restrict__ batch,
                       const float* __restrict__ ews,
                       const __bf16* __restrict__ WbT,
                       float* __restrict__ out, int N) {
    __shared__ __attribute__((aligned(16))) __bf16 xs[ROWS * XSTRIDE];
    __shared__ float es[ROWS * 8];
    __shared__ int   bs[ROWS];

    int tid = threadIdx.x;
    int lane = tid & 31, w = tid >> 5;
    int row0 = blockIdx.x * ROWS;

    // Stage x tile (f32 -> bf16), e tile, batch tile.
    for (int i = tid; i < ROWS * 64; i += 256) {      // one float4 per i
        int r = i >> 6, c4 = i & 63;
        int n = row0 + r;
        int nn = (n < N) ? n : N - 1;
        float4 xv = ((const float4*)x)[(size_t)nn * 64 + c4];
        if (n >= N) { xv.x = xv.y = xv.z = xv.w = 0.0f; }
        __bf16* dst = &xs[r * XSTRIDE + (c4 << 2)];
        dst[0] = (__bf16)xv.x; dst[1] = (__bf16)xv.y;
        dst[2] = (__bf16)xv.z; dst[3] = (__bf16)xv.w;
    }
    for (int i = tid; i < ROWS * 8; i += 256) {
        int n = row0 + (i >> 3);
        es[i] = (n < N) ? ews[(size_t)n * 8 + (i & 7)] : 0.0f;
    }
    for (int i = tid; i < ROWS; i += 256) {
        int n = row0 + i;
        bs[i] = (int)batch[(n < N) ? n : (N - 1)];
    }
    __syncthreads();

    int lane15 = lane & 15;
    int hi = (lane & 16) ? 1 : 0;                 // lane half
    #pragma unroll
    for (int cti = 0; cti < 2; ++cti) {
        int ct = w + cti * 8;                     // column tile 0..15
        int h  = ct >> 1;                         // head for this tile
        int col = ct * 16 + lane15;
        v8f acc0 = {0,0,0,0,0,0,0,0};
        v8f acc1 = acc0, acc2 = acc0, acc3 = acc0;

        const __bf16* bbase = WbT + (size_t)col * 256 + hi * 16;
        #pragma unroll
        for (int k = 0; k < 8; ++k) {
            Frag bf;
            const __bf16* bp = bbase + k * 32;
            bf.q[0] = *(const uint4*)bp;
            bf.q[1] = *(const uint4*)(bp + 8);
            #pragma unroll
            for (int rt = 0; rt < 4; ++rt) {
                Frag af;
                const __bf16* ap = &xs[(rt * 16 + lane15) * XSTRIDE + k * 32 + hi * 16];
                af.q[0] = *(const uint4*)ap;
                af.q[1] = *(const uint4*)(ap + 8);
                v8f c = (rt == 0) ? acc0 : (rt == 1) ? acc1 : (rt == 2) ? acc2 : acc3;
                c = __builtin_amdgcn_wmma_f32_16x16x32_bf16(
                        false, af.v, false, bf.v, (short)0, c, false, false);
                if (rt == 0) acc0 = c; else if (rt == 1) acc1 = c;
                else if (rt == 2) acc2 = c; else acc3 = c;
            }
        }

        // Epilogue: scale by e, reduce sorted-segment runs, one atomic/col.
        int mrow = hi * 8;
        #pragma unroll
        for (int rt = 0; rt < 4; ++rt) {
            v8f a = (rt == 0) ? acc0 : (rt == 1) ? acc1 : (rt == 2) ? acc2 : acc3;
            int rbase = rt * 16;
            #pragma unroll
            for (int j = 0; j < 8; ++j)
                a[j] *= es[(rbase + mrow + j) * 8 + h];

            int ss = 0;
            while (ss < 16) {
                int g = bs[rbase + ss];
                int se = ss + 1;
                while (se < 16 && bs[rbase + se] == g) ++se;
                float cs = 0.0f;
                #pragma unroll
                for (int j = 0; j < 8; ++j) {
                    int M = mrow + j;
                    cs += (M >= ss && M < se) ? a[j] : 0.0f;
                }
                cs += __shfl_xor(cs, 16, 32);
                if (lane < 16)
                    atomicAdd(&out[(size_t)g * 256 + ct * 16 + lane], cs);
                ss = se;
            }
        }
    }
}

// ---------------- normalize by s, add bias (sum(alpha)==1) --------------
__global__ void norm_k(float* __restrict__ out, const float* __restrict__ s,
                       const float* __restrict__ bn, int G) {
    int idx = blockIdx.x * 256 + threadIdx.x;
    if (idx >= G * 256) return;
    int g = idx >> 8;
    int h = (idx & 255) >> 5;
    float sv = s[g * 8 + h];
    out[idx] = (sv > 0.0f) ? (out[idx] / sv + bn[idx & 255]) : 0.0f;
}

extern "C" void kernel_launch(void* const* d_in, const int* in_sizes, int n_in,
                              void* d_out, int out_size, void* d_ws, size_t ws_size,
                              hipStream_t stream) {
    (void)n_in; (void)ws_size;
    const float*     x     = (const float*)d_in[0];
    const long long* batch = (const long long*)d_in[1];   // int64 per reference
    const float*     Wg    = (const float*)d_in[2];
    const float*     Wn    = (const float*)d_in[3];
    const float*     bn    = (const float*)d_in[4];
    float* out = (float*)d_out;

    int N = in_sizes[1];            // 500000
    int G = out_size / 256;         // 4096

    // Workspace layout: gate/e [N*8] | m [G*8] | s [G*8] | WbT bf16 [256*256]
    float*  gate = (float*)d_ws;
    float*  m    = gate + (size_t)N * 8;
    float*  s    = m + (size_t)G * 8;
    __bf16* WbT  = (__bf16*)(s + (size_t)G * 8);

    init_k<<<(G * 256 + 255) / 256, 256, 0, stream>>>(out, m, s, G);
    wcvt_k<<<256, 256, 0, stream>>>(Wn, WbT);
    gate_k<<<(N + 31) / 32, 256, 0, stream>>>(x, batch, Wg, gate, m, N);
    exp_k<<<(N * 8 + 255) / 256, 256, 0, stream>>>(batch, gate, m, s, N);
    main_k<<<(N + ROWS - 1) / ROWS, 256, 0, stream>>>(x, batch, gate, WbT, out, N);
    norm_k<<<(G * 256 + 255) / 256, 256, 0, stream>>>(out, s, bn, G);
}
